// DMLNegHead_59803124629819
// MI455X (gfx1250) — matrix operand order repaired and checked by reading, not compiled, not run on hardware
//
#include <hip/hip_runtime.h>
#include <hip/hip_bf16.h>
#include <math.h>

// ---------------------------------------------------------------------------
// DMLNegHead for MI455X (gfx1250): bf16 WMMA GEMM pipeline + fused epilogue.
//   emb  = W_emb @ x[b]          (GEMM1, bf16 wmma, f32 acc)
//   embN = l2norm(emb)           (wave-per-pixel reduction)
//   dot  = embN @ reps_packed^T  (GEMM2, bf16 wmma) -> fused distance/exp
//   cls_score normalized in a final coalesced pass (no float atomics).
// ---------------------------------------------------------------------------

typedef __attribute__((ext_vector_type(16))) __bf16 bf16x16;
typedef __attribute__((ext_vector_type(8)))  __bf16 bf16x8;
typedef __attribute__((ext_vector_type(8)))  float  f32x8;

namespace {
constexpr int    kB   = 32;
constexpr int    kC   = 256;    // classes == input channels
constexpr int    kHW  = 1024;   // 32*32 pixels
constexpr int    kE   = 512;    // embedding dim
constexpr int    kCM  = 1024;   // 256 classes * 4 packed modes (1 pos + 3 neg)
constexpr size_t kS1  = (size_t)kB * kC * kHW;  // 8388608 elements per output
}

// round-to-nearest-even f32 -> bf16
__device__ __forceinline__ __bf16 f2bf(float f) {
  union { float f; unsigned u; } v; v.f = f;
  unsigned r = v.u + 0x7FFFu + ((v.u >> 16) & 1u);
  unsigned short h = (unsigned short)(r >> 16);
  return __builtin_bit_cast(__bf16, h);
}

// A-tile (16x32, M x K, row-major source, ld in elements).
// Contiguous-k layout used here (same as loadB): lanes 0-15 -> M=lane, K 0..15;
//             lanes 16-31 -> M=lane-16, K 16..31.
__device__ __forceinline__ bf16x16 loadA(const __bf16* base, int ld, int lane) {
  int row = lane & 15;
  int kof = (lane >> 4) << 4;
  const __bf16* p = base + (size_t)row * ld + kof;
  bf16x8 lo = *(const bf16x8*)p;          // K = kof .. kof+7
  bf16x8 hi = *(const bf16x8*)(p + 8);    // K = kof+8 .. kof+15
  bf16x16 r;
#pragma unroll
  for (int i = 0; i < 8; ++i) { r[i] = lo[i]; r[i + 8] = hi[i]; }
  return r;
}

// B-tile (32x16, K x N). Source stored [N][K] row-major (column-major K x N),
// so each lane's 16 contiguous K values are one 32-byte load.
// lanes 0-15 -> N=lane, K 0..15; lanes 16-31 -> N=lane-16, K 16..31.
__device__ __forceinline__ bf16x16 loadB(const __bf16* base, int ld, int lane) {
  int col = lane & 15;
  int kof = (lane >> 4) << 4;
  return *(const bf16x16*)(base + (size_t)col * ld + kof);
}

__device__ __forceinline__ f32x8 wmma_bf16(bf16x16 a, bf16x16 b, f32x8 c) {
  return __builtin_amdgcn_wmma_f32_16x16x32_bf16(
      /*neg_a=*/false, a, /*neg_b=*/false, b,
      /*c_mod=*/(short)0, c, /*reuse_a=*/false, /*reuse_b=*/false);
}

// Store D tile into an [N][M] f32 buffer (out at tile origin, ld in elements).
// VGPR r, lane<16: (M=r, N=lane); lane>=16: (M=8+r, N=lane-16).
__device__ __forceinline__ void storeD(float* out, int ld, int lane, f32x8 acc) {
  int n  = lane & 15;
  int mb = (lane >> 4) << 3;
  float* p = out + (size_t)n * ld + mb;
  float4 lo; lo.x = acc[0]; lo.y = acc[1]; lo.z = acc[2]; lo.w = acc[3];
  float4 hi; hi.x = acc[4]; hi.y = acc[5]; hi.z = acc[6]; hi.w = acc[7];
  *(float4*)p       = lo;
  *((float4*)p + 1) = hi;
}

// ---------------------------------------------------------------------------
// Kernel 1: build packed, L2-normalized representative matrix
// repsN[c][m][e], m=0 positive mode, m=1..3 negative modes. One wave per row.
// ---------------------------------------------------------------------------
__global__ __launch_bounds__(256) void prep_reps_kernel(
    const float* __restrict__ rep_w,  const float* __restrict__ rep_b,
    const float* __restrict__ nrep_w, const float* __restrict__ nrep_b,
    __bf16* __restrict__ repsN) {
  int row  = (blockIdx.x * blockDim.x + threadIdx.x) >> 5;  // 0..1023
  int lane = threadIdx.x & 31;
  int c = row >> 2, m = row & 3;
  const float* ws; const float* bs; size_t base;
  if (m == 0) { ws = rep_w;  bs = rep_b;  base = (size_t)c * kE; }
  else        { ws = nrep_w; bs = nrep_b; base = ((size_t)c * 3 + (m - 1)) * kE; }
  float v[16]; float ss = 0.f;
#pragma unroll
  for (int j = 0; j < 16; ++j) {
    float t = ws[base + lane + j * 32] + bs[base + lane + j * 32];
    v[j] = t; ss += t * t;
  }
#pragma unroll
  for (int off = 16; off >= 1; off >>= 1) ss += __shfl_xor(ss, off, 32);
  float scale = 1.0f / fmaxf(sqrtf(ss), 1e-12f);
  __bf16* dst = repsN + (size_t)row * kE;
#pragma unroll
  for (int j = 0; j < 16; ++j) dst[lane + j * 32] = f2bf(v[j] * scale);
}

// ---------------------------------------------------------------------------
// Kernel 2: W_emb f32 -> bf16 (layout unchanged: [E][C], A-matrix of GEMM1)
// ---------------------------------------------------------------------------
__global__ __launch_bounds__(256) void conv_w_kernel(
    const float* __restrict__ W, __bf16* __restrict__ Wb) {
  int i = blockIdx.x * blockDim.x + threadIdx.x;   // 512*256 = 131072 total
  Wb[i] = f2bf(W[i]);
}

// ---------------------------------------------------------------------------
// Kernel 3: x [B][C][HW] f32 -> xT [B][HW][C] bf16 (LDS-tiled transpose)
// ---------------------------------------------------------------------------
__global__ __launch_bounds__(256) void transpose_x_kernel(
    const float* __restrict__ x, __bf16* __restrict__ xT) {
  __shared__ __bf16 tile[32][33];
  int b  = blockIdx.z;
  int p0 = blockIdx.x * 32;
  int c0 = blockIdx.y * 32;
  int tx = threadIdx.x, ty = threadIdx.y;          // 32 x 8
#pragma unroll
  for (int k = 0; k < 4; ++k) {
    int c = c0 + ty + k * 8;
    tile[ty + k * 8][tx] = f2bf(x[((size_t)b * kC + c) * kHW + p0 + tx]);
  }
  __syncthreads();
#pragma unroll
  for (int k = 0; k < 4; ++k) {
    int p = p0 + ty + k * 8;
    xT[((size_t)b * kHW + p) * kC + c0 + tx] = tile[tx][ty + k * 8];
  }
}

// ---------------------------------------------------------------------------
// Kernel 4: GEMM1  embT[b][p][e] = sum_c W_emb[e][c] * x[b][c][p]
// One wave computes a 32(E) x 32(pixel) tile; K=256 in steps of 32.
// ---------------------------------------------------------------------------
__global__ __launch_bounds__(128) void gemm1_kernel(
    const __bf16* __restrict__ Wb,   // [512][256]
    const __bf16* __restrict__ xT,   // [B][1024][256]
    float* __restrict__ embT) {      // [B][1024][512]  ([N][M])
  int lane = threadIdx.x & 31;
  int wid  = (blockIdx.x * blockDim.x + threadIdx.x) >> 5;
  int b    = wid >> 9;          // 512 tiles per batch
  int rem  = wid & 511;
  int mt   = rem >> 5;          // 16 E-tiles
  int nt   = rem & 31;          // 32 pixel-tiles
  int m0 = mt * 32, n0 = nt * 32;
  const __bf16* Ab = Wb;
  const __bf16* Bb = xT + (size_t)b * kHW * kC;
  f32x8 acc00 = {}, acc01 = {}, acc10 = {}, acc11 = {};
#pragma unroll 2
  for (int k0 = 0; k0 < kC; k0 += 32) {
    bf16x16 a0 = loadA(Ab + (size_t)m0 * kC + k0,        kC, lane);
    bf16x16 a1 = loadA(Ab + (size_t)(m0 + 16) * kC + k0, kC, lane);
    bf16x16 b0 = loadB(Bb + (size_t)n0 * kC + k0,        kC, lane);
    bf16x16 b1 = loadB(Bb + (size_t)(n0 + 16) * kC + k0, kC, lane);
    acc00 = wmma_bf16(a0, b0, acc00);
    acc01 = wmma_bf16(a0, b1, acc01);
    acc10 = wmma_bf16(a1, b0, acc10);
    acc11 = wmma_bf16(a1, b1, acc11);
  }
  float* Ob = embT + (size_t)b * kHW * kE;
  storeD(Ob + (size_t)n0 * kE        + m0,      kE, lane, acc00);
  storeD(Ob + (size_t)(n0 + 16) * kE + m0,      kE, lane, acc01);
  storeD(Ob + (size_t)n0 * kE        + m0 + 16, kE, lane, acc10);
  storeD(Ob + (size_t)(n0 + 16) * kE + m0 + 16, kE, lane, acc11);
}

// ---------------------------------------------------------------------------
// Kernel 5: per-pixel L2 normalization over E=512; f32 -> bf16. Wave per pixel.
// ---------------------------------------------------------------------------
__global__ __launch_bounds__(256) void norm_emb_kernel(
    const float* __restrict__ embT, __bf16* __restrict__ embN) {
  int wid  = (blockIdx.x * blockDim.x + threadIdx.x) >> 5;  // 0..32767 (b*HW+p)
  int lane = threadIdx.x & 31;
  const float* src = embT + (size_t)wid * kE;
  float v[16]; float ss = 0.f;
#pragma unroll
  for (int j = 0; j < 16; ++j) { float t = src[lane + j * 32]; v[j] = t; ss += t * t; }
#pragma unroll
  for (int off = 16; off >= 1; off >>= 1) ss += __shfl_xor(ss, off, 32);
  float scale = 1.0f / fmaxf(sqrtf(ss), 1e-12f);
  __bf16* dst = embN + (size_t)wid * kE;
#pragma unroll
  for (int j = 0; j < 16; ++j) dst[lane + j * 32] = f2bf(v[j] * scale);
}

// ---------------------------------------------------------------------------
// Kernel 6: GEMM2 + fused epilogue.
// Block (128 thr, 4 waves) computes dot tile: 64 pixels x 64 cm (16 classes x
// 4 modes) into LDS, then each thread computes distances / exp outputs for
// (pixel, class) pairs and writes all 5 output tensors (cls_score unnormalized).
// ---------------------------------------------------------------------------
__global__ __launch_bounds__(128) void gemm2_kernel(
    const __bf16* __restrict__ embN,   // [B][1024][512]
    const __bf16* __restrict__ repsN,  // [1024 cm][512]
    float* __restrict__ out) {
  __shared__ float dot_s[64][65];
  int lane = threadIdx.x & 31;
  int w    = threadIdx.x >> 5;
  int blk  = blockIdx.x;                 // 32 * 16 * 16 = 8192 blocks
  int b    = blk >> 8;
  int pt   = (blk >> 4) & 15;
  int ct   = blk & 15;
  int p0   = pt * 64;
  int cm0  = ct * 64;
  int waveM = (w >> 1) * 32;             // pixel offset inside block tile
  int waveN = (w & 1) * 32;              // cm offset inside block tile
  const __bf16* Ab = embN + ((size_t)b * kHW + p0 + waveM) * kE;
  const __bf16* Bb = repsN + (size_t)(cm0 + waveN) * kE;
  f32x8 acc00 = {}, acc01 = {}, acc10 = {}, acc11 = {};
#pragma unroll 2
  for (int k0 = 0; k0 < kE; k0 += 32) {
    bf16x16 a0 = loadA(Ab + k0,                    kE, lane);
    bf16x16 a1 = loadA(Ab + (size_t)16 * kE + k0,  kE, lane);
    bf16x16 b0 = loadB(Bb + k0,                    kE, lane);
    bf16x16 b1 = loadB(Bb + (size_t)16 * kE + k0,  kE, lane);
    acc00 = wmma_bf16(a0, b0, acc00);
    acc01 = wmma_bf16(a0, b1, acc01);
    acc10 = wmma_bf16(a1, b0, acc10);
    acc11 = wmma_bf16(a1, b1, acc11);
  }
  // D -> LDS  (dot_s[pixel][cm])
  int nl = lane & 15;
  int mb = (lane >> 4) << 3;
#pragma unroll
  for (int r = 0; r < 8; ++r) {
    dot_s[waveM + mb + r     ][waveN + nl     ] = acc00[r];
    dot_s[waveM + mb + r     ][waveN + nl + 16] = acc01[r];
    dot_s[waveM + mb + r + 16][waveN + nl     ] = acc10[r];
    dot_s[waveM + mb + r + 16][waveN + nl + 16] = acc11[r];
  }
  __syncthreads();

  const float inv2s2 = 2.0f;             // 1 / (2 * 0.5^2)
  int cbase = cm0 >> 2;                  // first class handled by this block
#pragma unroll
  for (int it = 0; it < 8; ++it) {
    int id  = threadIdx.x + (it << 7);   // 0..1023 = 64 pixels x 16 classes
    int pix = id & 63;
    int cls = id >> 6;
    float q0 = fmaxf(2.f - 2.f * dot_s[pix][cls * 4 + 0], 0.f);
    float q1 = fmaxf(2.f - 2.f * dot_s[pix][cls * 4 + 1], 0.f);
    float q2 = fmaxf(2.f - 2.f * dot_s[pix][cls * 4 + 2], 0.f);
    float q3 = fmaxf(2.f - 2.f * dot_s[pix][cls * 4 + 3], 0.f);
    float dpos = sqrtf(q0);
    float d1 = sqrtf(q1), d2 = sqrtf(q2), d3 = sqrtf(q3);
    float pneg = fmaxf(fmaxf(__expf(-q1 * inv2s2), __expf(-q2 * inv2s2)),
                       __expf(-q3 * inv2s2));
    float negmin = fminf(fminf(d1, d2), d3);
    float t = dpos + 0.3f * fmaxf(2.0f - negmin, 0.f);   // BETA, NEG_SCOPE
    float probs = __expf(-t * t * inv2s2);
    int c = cbase + cls;
    int p = p0 + pix;
    size_t base = ((size_t)(b * kC + c)) * kHW + p;
    out[base]            = probs;               // cls_score (unnormalized)
    out[kS1 + base]      = pneg;                // cls_score_neg
    out[2 * kS1 + base]  = dpos;                // distances
    size_t nb = ((size_t)(b * kC + c) * 3) * kHW + p;
    out[3 * kS1 + nb]            = d1;          // distances_neg m=0..2
    out[3 * kS1 + nb + kHW]      = d2;
    out[3 * kS1 + nb + 2 * kHW]  = d3;
    out[6 * kS1 + base]  = __expf(-q0 * inv2s2);// probs_ori
  }
}

// ---------------------------------------------------------------------------
// Kernel 7: cls_score normalization: divide by sum over classes per pixel.
// Thread = pixel; class loop reads/writes are coalesced across threads.
// ---------------------------------------------------------------------------
__global__ __launch_bounds__(256) void cls_norm_kernel(float* __restrict__ out) {
  int b  = blockIdx.x >> 2;              // 32 batches x 4 pixel chunks
  int p  = ((blockIdx.x & 3) << 8) + threadIdx.x;
  size_t base = ((size_t)b * kC) * kHW + p;
  float s = 0.f;
  for (int c = 0; c < kC; ++c) s += out[base + (size_t)c * kHW];
  float inv = 1.0f / s;
  for (int c = 0; c < kC; ++c) out[base + (size_t)c * kHW] *= inv;
}

// ---------------------------------------------------------------------------
extern "C" void kernel_launch(void* const* d_in, const int* in_sizes, int n_in,
                              void* d_out, int out_size, void* d_ws, size_t ws_size,
                              hipStream_t stream) {
  (void)in_sizes; (void)n_in; (void)out_size; (void)ws_size;
  const float* x      = (const float*)d_in[0];
  const float* W_emb  = (const float*)d_in[1];
  const float* rep_w  = (const float*)d_in[2];
  const float* rep_b  = (const float*)d_in[3];
  const float* nrep_w = (const float*)d_in[4];
  const float* nrep_b = (const float*)d_in[5];
  float* out = (float*)d_out;

  // Workspace carve-up (256B-aligned regions, ~113.25 MB total):
  char* ws = (char*)d_ws;
  __bf16* repsN = (__bf16*)(ws);                     // 1024*512*2      = 1 MB
  __bf16* Wb    = (__bf16*)(ws + 0x00100000);        // 512*256*2       = 256 KB
  __bf16* xT    = (__bf16*)(ws + 0x00140000);        // 32*1024*256*2   = 16 MB
  float*  embT  = (float*) (ws + 0x01140000);        // 32*1024*512*4   = 64 MB
  __bf16* embN  = (__bf16*)(ws + 0x05140000);        // 32*1024*512*2   = 32 MB

  prep_reps_kernel  <<<dim3(128),       dim3(256),    0, stream>>>(rep_w, rep_b, nrep_w, nrep_b, repsN);
  conv_w_kernel     <<<dim3(512),       dim3(256),    0, stream>>>(W_emb, Wb);
  transpose_x_kernel<<<dim3(32, 8, 32), dim3(32, 8),  0, stream>>>(x, xT);
  gemm1_kernel      <<<dim3(4096),      dim3(128),    0, stream>>>(Wb, xT, embT);
  norm_emb_kernel   <<<dim3(4096),      dim3(256),    0, stream>>>(embT, embN);
  gemm2_kernel      <<<dim3(8192),      dim3(128),    0, stream>>>(embN, repsN, out);
  cls_norm_kernel   <<<dim3(128),       dim3(256),    0, stream>>>(out);
}
